// LSTM_Net_36747740185104
// MI455X (gfx1250) — matrix operand order repaired and checked by reading, not compiled, not run on hardware
//
#include <hip/hip_runtime.h>

// ---------- types ----------
typedef __attribute__((ext_vector_type(8)))  float          v8f;
typedef __attribute__((ext_vector_type(16))) __bf16         bf16x16;
typedef __attribute__((ext_vector_type(4)))  unsigned int   u32x4;
typedef __attribute__((ext_vector_type(8)))  unsigned int   u32x8;

struct Q2 { u32x4 a, b; };   // 32 bytes -> bit_cast to bf16x16

// ---------- problem sizes ----------
#define BB 256
#define TT 336
#define II 16
#define HH 512
#define LL 256
#define OO 10
#define G4 2048   // 4*H

// ---------- workspace layout (bytes) ----------
// packed Whh (+Wih as k-tile 16): 17 ktiles * 128 ntiles * 32 lanes * 32 B
#define OFF_PW    0
#define SZ_PW     (17*128*32*32)          // 2,228,224
#define OFF_PBLW  (OFF_PW + SZ_PW)
#define SZ_PBLW   (16*16*32*32)           // 262,144
#define OFF_BS    (OFF_PBLW + SZ_PBLW)
#define SZ_BS     (G4*4)
#define OFF_BLB   (OFF_BS + SZ_BS)
#define SZ_BLB    (LL*4)
#define OFF_LAST  (OFF_BLB + SZ_BLB)
#define SZ_LAST   (256*HH*4)              // 524,288
#define OFF_Y     (OFF_LAST + SZ_LAST)
#define SZ_Y      (256*LL*4)              // 262,144

// ---------- helpers ----------
__device__ __forceinline__ float softplusf(float x) {
    return (x > 20.f) ? x : log1pf(expf(x));
}
__device__ __forceinline__ unsigned f2bf1(float f) {
    unsigned u = __float_as_uint(f);
    return (u + 0x7FFFu + ((u >> 16) & 1u)) >> 16;   // RNE bf16
}
__device__ __forceinline__ unsigned packbf(float lo, float hi) {
    return f2bf1(lo) | (f2bf1(hi) << 16);
}
__device__ __forceinline__ float sigm(float x) { return 1.f / (1.f + expf(-x)); }

// ==========================================================================
// 1) sample bias vectors: b (4H) and BLb (L)
// ==========================================================================
__global__ void sample_vec_kernel(const float* __restrict__ b_mu,  const float* __restrict__ b_rho,  const float* __restrict__ eps_b,
                                  const float* __restrict__ blb_mu,const float* __restrict__ blb_rho,const float* __restrict__ eps_blb,
                                  float* __restrict__ bS, float* __restrict__ blbS) {
    int i = blockIdx.x * 256 + threadIdx.x;
    if (i < G4) {
        bS[i] = b_mu[i] + softplusf(b_rho[i]) * eps_b[i];
    } else if (i < G4 + LL) {
        int j = i - G4;
        blbS[j] = blb_mu[j] + softplusf(blb_rho[j]) * eps_blb[j];
    }
}

// ==========================================================================
// 2) sample Whh/Wih and pack into WMMA B-fragment layout (bf16).
//    B (32x16, 16-bit): lane L holds column N=L%16; half=L/16 covers
//    K = half*8 .. half*8+7 and 16+half*8 .. 16+half*8+7 (ISA 7.12.4/7.12.2).
//    k-tiles 0..15: Whh rows kt*32+k ; k-tile 16: Wih rows 0..15, rows 16..31 zero.
// ==========================================================================
__global__ void pack_whh_kernel(const float* __restrict__ whh_mu, const float* __restrict__ whh_rho, const float* __restrict__ eps_whh,
                                const float* __restrict__ wih_mu, const float* __restrict__ wih_rho, const float* __restrict__ eps_wih,
                                unsigned* __restrict__ pw) {
    int g    = blockIdx.x * 256 + threadIdx.x;      // 0 .. 17*128*32-1
    int lane = g & 31;
    int nt   = (g >> 5) & 127;
    int kt   = g >> 12;
    if (kt >= 17) return;
    int N16 = lane & 15, half = lane >> 4;
    int gc  = nt * 16 + N16;                        // gate column 0..2047
    unsigned au[8];
#pragma unroll
    for (int j = 0; j < 8; ++j) {
        int k0 = half * 8 + ((j < 4) ? 2 * j : 16 + 2 * (j - 4));   // K, K+1 within this 32-chunk
        float v0, v1;
        if (kt < 16) {
            int r0 = kt * 32 + k0;
            v0 = whh_mu[(size_t)r0 * G4 + gc] + softplusf(whh_rho[(size_t)r0 * G4 + gc]) * eps_whh[(size_t)r0 * G4 + gc];
            v1 = whh_mu[(size_t)(r0+1) * G4 + gc] + softplusf(whh_rho[(size_t)(r0+1) * G4 + gc]) * eps_whh[(size_t)(r0+1) * G4 + gc];
        } else {
            v0 = (k0     < II) ? (wih_mu[(size_t)k0 * G4 + gc]     + softplusf(wih_rho[(size_t)k0 * G4 + gc])     * eps_wih[(size_t)k0 * G4 + gc])     : 0.f;
            v1 = (k0 + 1 < II) ? (wih_mu[(size_t)(k0+1) * G4 + gc] + softplusf(wih_rho[(size_t)(k0+1) * G4 + gc]) * eps_wih[(size_t)(k0+1) * G4 + gc]) : 0.f;
        }
        au[j] = packbf(v0, v1);
    }
    unsigned* dst = pw + (((kt * 128 + nt) * 32 + lane) << 3);
    u32x4 lo = {au[0], au[1], au[2], au[3]};
    u32x4 hi = {au[4], au[5], au[6], au[7]};
    *reinterpret_cast<u32x4*>(dst)     = lo;
    *reinterpret_cast<u32x4*>(dst + 4) = hi;
}

// ==========================================================================
// 3) sample BLW [L,H] and pack as B-fragments for y = lastd @ BLW^T
//    B element (k, n) = BLW[n*H + k];  16 ktiles x 16 ntiles.
// ==========================================================================
__global__ void pack_blw_kernel(const float* __restrict__ blw_mu, const float* __restrict__ blw_rho, const float* __restrict__ eps_blw,
                                unsigned* __restrict__ pblw) {
    int g    = blockIdx.x * 256 + threadIdx.x;      // 0 .. 16*16*32-1
    int lane = g & 31;
    int nt   = (g >> 5) & 15;
    int kt   = g >> 9;
    if (kt >= 16) return;
    int N16 = lane & 15, half = lane >> 4;
    int n   = nt * 16 + N16;
    unsigned au[8];
#pragma unroll
    for (int j = 0; j < 8; ++j) {
        int k0 = kt * 32 + half * 8 + ((j < 4) ? 2 * j : 16 + 2 * (j - 4));
        size_t i0 = (size_t)n * HH + k0;
        float v0 = blw_mu[i0]     + softplusf(blw_rho[i0])     * eps_blw[i0];
        float v1 = blw_mu[i0 + 1] + softplusf(blw_rho[i0 + 1]) * eps_blw[i0 + 1];
        au[j] = packbf(v0, v1);
    }
    unsigned* dst = pblw + (((kt * 16 + nt) * 32 + lane) << 3);
    u32x4 lo = {au[0], au[1], au[2], au[3]};
    u32x4 hi = {au[4], au[5], au[6], au[7]};
    *reinterpret_cast<u32x4*>(dst)     = lo;
    *reinterpret_cast<u32x4*>(dst + 4) = hi;
}

// ==========================================================================
// 4) LSTM recurrence — only batch row 255 affects the output (reshape bug),
//    so ONE workgroup (16 waves) runs the whole scan. M=16 WMMA tile with
//    rows 0..14 zero. h: bf16 in LDS; c: fp32 in VGPRs; Whh: packed, L2.
//    gates[16,2048] per step = 16 waves x 8 n-tiles x 17 k-chunks WMMA.
// ==========================================================================
__global__ void __launch_bounds__(512)
lstm_kernel(const float* __restrict__ x, const float* __restrict__ dx,
            const unsigned* __restrict__ pw, const float* __restrict__ bS,
            float* __restrict__ lastT) {
    __shared__ unsigned short hsh[16 * HH];   // 16 KB, bf16 [row][k]
    __shared__ unsigned short xsh[16 * 32];   // 1 KB,  bf16 [row][k], cols 16..31 zero

    const int tid  = threadIdx.x;
    const int lane = tid & 31;
    const int wave = tid >> 5;
    const int N16  = lane & 15;
    const int half = lane >> 4;
    const int hct0 = wave * 2;                // this wave owns h-col tiles hct0, hct0+1

    for (int i = tid; i < 16 * HH; i += 512) hsh[i] = 0;
    for (int i = tid; i < 16 * 32; i += 512) xsh[i] = 0;
    __syncthreads();

    const v8f vzero = {0.f, 0.f, 0.f, 0.f, 0.f, 0.f, 0.f, 0.f};
    v8f creg[2] = {vzero, vzero};

    for (int t = 0; t < TT; ++t) {
        // xd tile: only row 15 (batch 255) is live
        if (tid < II) {
            int idx = (255 * TT + t) * II + tid;
            xsh[15 * 32 + tid] = (unsigned short)f2bf1(x[idx] * dx[idx]);
        }
        __syncthreads();

        v8f acc[8];
#pragma unroll
        for (int i = 0; i < 8; ++i) acc[i] = vzero;

        auto mm8 = [&](bf16x16 av, int kt) {
#pragma unroll
            for (int hi2 = 0; hi2 < 2; ++hi2) {
#pragma unroll
                for (int q = 0; q < 4; ++q) {
                    int nt = q * 32 + hct0 + hi2;
                    const u32x8* bp = reinterpret_cast<const u32x8*>(pw + (((kt * 128 + nt) * 32 + lane) << 3));
                    bf16x16 bv = __builtin_bit_cast(bf16x16, *bp);
                    acc[hi2 * 4 + q] = __builtin_amdgcn_wmma_f32_16x16x32_bf16(
                        false, av, false, bv, (short)0, acc[hi2 * 4 + q], false, false);
                }
            }
        };

        // A fragment (16-bit A 16x32, ISA 7.12.2): lane row M=N16,
        // VGPR0-3: K=half*8+0..7 ; VGPR4-7: K=16+half*8+0..7  -> 2x b128 LDS loads
        for (int kt = 0; kt < 16; ++kt) {
            const unsigned short* ap = hsh + N16 * HH + kt * 32 + half * 8;
            Q2 q2;
            q2.a = *reinterpret_cast<const u32x4*>(ap);
            q2.b = *reinterpret_cast<const u32x4*>(ap + 16);
            mm8(__builtin_bit_cast(bf16x16, q2), kt);
        }
        {   // k-chunk 16: input projection (Wih), K padded 16->32 with zeros
            const unsigned short* ap = xsh + N16 * 32 + half * 8;
            Q2 q2;
            q2.a = *reinterpret_cast<const u32x4*>(ap);
            q2.b = *reinterpret_cast<const u32x4*>(ap + 16);
            mm8(__builtin_bit_cast(bf16x16, q2), 16);
        }

        // pointwise gates -> c, h   (D layout: elem r -> row M=r+8*half, col N16)
        float hval[2][8];
#pragma unroll
        for (int hi2 = 0; hi2 < 2; ++hi2) {
            int hcol = (hct0 + hi2) * 16 + N16;
            float bi = bS[hcol], bf_ = bS[HH + hcol], bg = bS[2 * HH + hcol], bo = bS[3 * HH + hcol];
#pragma unroll
            for (int r = 0; r < 8; ++r) {
                float ig = sigm(acc[hi2 * 4 + 0][r] + bi);
                float fg = sigm(acc[hi2 * 4 + 1][r] + bf_);
                float gg = tanhf(acc[hi2 * 4 + 2][r] + bg);
                float og = sigm(acc[hi2 * 4 + 3][r] + bo);
                float cc = fg * creg[hi2][r] + ig * gg;
                creg[hi2][r] = cc;
                hval[hi2][r] = og * tanhf(cc);
            }
        }
        __syncthreads();   // all hsh reads of this step done

        // write back only the live row (M=15 -> r=7, half=1)
        if (half == 1) {
#pragma unroll
            for (int hi2 = 0; hi2 < 2; ++hi2) {
                int col = (hct0 + hi2) * 16 + N16;
                hsh[15 * HH + col] = (unsigned short)f2bf1(hval[hi2][7]);
                if (t >= 80) lastT[(t - 80) * HH + col] = hval[hi2][7];
            }
        }
        __syncthreads();
    }
}

// ==========================================================================
// 5) head: y = relu((last*drop_h) @ BLW^T + BLb) * drop_l   -> ybuf [256,256]
//    256 tiles of 16x16, K=512, bf16 WMMA. 32 blocks x 8 waves.
// ==========================================================================
__global__ void __launch_bounds__(256)
head_kernel(const float* __restrict__ lastT, const float* __restrict__ dh,
            const unsigned* __restrict__ pblw, const float* __restrict__ blbS,
            const float* __restrict__ dl, float* __restrict__ ybuf) {
    const int lane = threadIdx.x & 31;
    const int wave = threadIdx.x >> 5;
    const int tile = blockIdx.x * 8 + wave;        // 0..255
    const int mt = tile >> 4, nt = tile & 15;
    const int N16 = lane & 15, half = lane >> 4;
    const int Mrow = mt * 16 + N16;
    const float* lr = lastT + (size_t)Mrow * HH;
    const float* dr = dh    + (size_t)Mrow * HH;

    const v8f vzero = {0.f, 0.f, 0.f, 0.f, 0.f, 0.f, 0.f, 0.f};
    v8f acc = vzero;

    for (int kt = 0; kt < 16; ++kt) {
        unsigned au[8];
#pragma unroll
        for (int j = 0; j < 8; ++j) {              // A layout (16-bit 16x32)
            int k0 = kt * 32 + half * 8 + ((j < 4) ? 2 * j : 16 + 2 * (j - 4));
            au[j] = packbf(lr[k0] * dr[k0], lr[k0 + 1] * dr[k0 + 1]);
        }
        Q2 qa;
        qa.a = (u32x4){au[0], au[1], au[2], au[3]};
        qa.b = (u32x4){au[4], au[5], au[6], au[7]};
        bf16x16 av = __builtin_bit_cast(bf16x16, qa);
        const u32x8* bp = reinterpret_cast<const u32x8*>(pblw + (((kt * 16 + nt) * 32 + lane) << 3));
        bf16x16 bv = __builtin_bit_cast(bf16x16, *bp);
        acc = __builtin_amdgcn_wmma_f32_16x16x32_bf16(false, av, false, bv, (short)0, acc, false, false);
    }

    int n = nt * 16 + N16;
    float bb = blbS[n];
#pragma unroll
    for (int r = 0; r < 8; ++r) {
        int m = mt * 16 + r + 8 * half;
        float v = acc[r] + bb;
        v = (v > 0.f) ? v : 0.f;
        ybuf[(size_t)m * LL + n] = v * dl[(size_t)m * LL + n];
    }
}

// ==========================================================================
// 6) out = yd @ lin_w^T   [256,256] x [10,256]^T -> [256,10]  (tiny, VALU)
// ==========================================================================
__global__ void out_kernel(const float* __restrict__ ybuf, const float* __restrict__ lw,
                           float* __restrict__ out) {
    int i = blockIdx.x * 256 + threadIdx.x;
    if (i >= 256 * OO) return;
    int m = i / OO, o = i % OO;
    const float* yr = ybuf + (size_t)m * LL;
    const float* wr = lw + (size_t)o * LL;
    float s = 0.f;
    for (int n = 0; n < LL; ++n) s += yr[n] * wr[n];
    out[i] = s;
}

// ==========================================================================
extern "C" void kernel_launch(void* const* d_in, const int* in_sizes, int n_in,
                              void* d_out, int out_size, void* d_ws, size_t ws_size,
                              hipStream_t stream) {
    const float* x       = (const float*)d_in[0];
    const float* drop_x  = (const float*)d_in[1];
    const float* drop_h  = (const float*)d_in[2];
    const float* drop_l  = (const float*)d_in[3];
    const float* wih_mu  = (const float*)d_in[4];
    const float* wih_rho = (const float*)d_in[5];
    const float* eps_wih = (const float*)d_in[6];
    const float* whh_mu  = (const float*)d_in[7];
    const float* whh_rho = (const float*)d_in[8];
    const float* eps_whh = (const float*)d_in[9];
    const float* b_mu    = (const float*)d_in[10];
    const float* b_rho   = (const float*)d_in[11];
    const float* eps_b   = (const float*)d_in[12];
    const float* blw_mu  = (const float*)d_in[13];
    const float* blw_rho = (const float*)d_in[14];
    const float* eps_blw = (const float*)d_in[15];
    const float* blb_mu  = (const float*)d_in[16];
    const float* blb_rho = (const float*)d_in[17];
    const float* eps_blb = (const float*)d_in[18];
    const float* lin_w   = (const float*)d_in[19];
    float* out = (float*)d_out;

    char* ws = (char*)d_ws;
    unsigned* pw   = (unsigned*)(ws + OFF_PW);
    unsigned* pblw = (unsigned*)(ws + OFF_PBLW);
    float*    bS   = (float*)(ws + OFF_BS);
    float*    blbS = (float*)(ws + OFF_BLB);
    float*    lastT= (float*)(ws + OFF_LAST);
    float*    ybuf = (float*)(ws + OFF_Y);

    sample_vec_kernel<<<(G4 + LL + 255) / 256, 256, 0, stream>>>(
        b_mu, b_rho, eps_b, blb_mu, blb_rho, eps_blb, bS, blbS);

    pack_whh_kernel<<<(17 * 128 * 32) / 256, 256, 0, stream>>>(
        whh_mu, whh_rho, eps_whh, wih_mu, wih_rho, eps_wih, pw);

    pack_blw_kernel<<<(16 * 16 * 32) / 256, 256, 0, stream>>>(
        blw_mu, blw_rho, eps_blw, pblw);

    lstm_kernel<<<1, 512, 0, stream>>>(x, drop_x, pw, bS, lastT);

    head_kernel<<<32, 256, 0, stream>>>(lastT, drop_h, pblw, blbS, drop_l, ybuf);

    out_kernel<<<10, 256, 0, stream>>>(ybuf, lin_w, out);
}
